// VAE_21792664060433
// MI455X (gfx1250) — matrix operand order repaired and checked by reading, not compiled, not run on hardware
//
#include <hip/hip_runtime.h>

#define N_NODES 50000
#define N_EDGES 800000
#define DIM_IN  256
#define DIM_HID 128
#define DIM_OUT 64

typedef __attribute__((ext_vector_type(16))) __bf16 v16bf;
typedef __attribute__((ext_vector_type(8)))  __bf16 v8bf;
typedef __attribute__((ext_vector_type(8)))  float  v8f;

// float -> bf16 with round-to-nearest-even (bit-level, no __bf16 arithmetic needed)
__device__ __forceinline__ __bf16 f2bf(float f) {
  unsigned u = __float_as_uint(f);
  u = (u + 0x7FFFu + ((u >> 16) & 1u)) >> 16;
  unsigned short s = (unsigned short)u;
  __bf16 r;
  __builtin_memcpy(&r, &s, sizeof(r));
  return r;
}

// ---------------- degree count: deg[dst] += 1 ----------------
__global__ void deg_kernel(const int* __restrict__ dst, float* __restrict__ deg, int E) {
  int e = blockIdx.x * blockDim.x + threadIdx.x;
  if (e < E) atomicAdd(&deg[dst[e]], 1.0f);
}

// ---------------- scatter-add: agg[dst,:] += F[src,:] (float4 granularity) ----------------
__global__ void scatter_kernel(const float* __restrict__ F, const int* __restrict__ src,
                               const int* __restrict__ dst, float* __restrict__ agg,
                               int E, int D) {
  int t = blockIdx.x * blockDim.x + threadIdx.x;
  int vec = D >> 2;
  int total = E * vec;              // max 51.2M, fits int
  if (t >= total) return;
  int e = t / vec;
  int g = t - e * vec;
  int s = src[e], d = dst[e];
  float4 v = ((const float4*)F)[(size_t)s * vec + g];
  float* o = agg + (size_t)d * D + (size_t)g * 4;
  atomicAdd(o + 0, v.x);
  atomicAdd(o + 1, v.y);
  atomicAdd(o + 2, v.z);
  atomicAdd(o + 3, v.w);
}

// ---------------- build bf16 A = [agg/max(deg,1) | F]  (N x 2D) ----------------
__global__ void build_a_kernel(const float* __restrict__ agg, const float* __restrict__ deg,
                               const float* __restrict__ F, __bf16* __restrict__ Ab,
                               int N, int D) {
  int i = blockIdx.x * blockDim.x + threadIdx.x;
  if (i >= N * D) return;
  int n = i / D;
  int f = i - n * D;
  float dg = fmaxf(deg[n], 1.0f);
  size_t base = (size_t)n * (size_t)(2 * D);
  Ab[base + f]     = f2bf(agg[i] / dg);
  Ab[base + D + f] = f2bf(F[i]);
}

// ---------------- plain f32 -> bf16 convert ----------------
__global__ void conv_bf_kernel(const float* __restrict__ F, __bf16* __restrict__ o, int total) {
  int i = blockIdx.x * blockDim.x + threadIdx.x;
  if (i < total) o[i] = f2bf(F[i]);
}

// ---------------- weight prep: Bt[c][k] = bf16( [Wa;Wb][k][c] ), row-major NOUT x (Ka+Kb) ----------------
__global__ void prep_w_kernel(const float* __restrict__ Wa, const float* __restrict__ Wb,
                              __bf16* __restrict__ Bt, int Ka, int Kb, int NOUT) {
  int Ktot = Ka + Kb;
  int i = blockIdx.x * blockDim.x + threadIdx.x;
  if (i >= NOUT * Ktot) return;
  int c = i / Ktot;
  int k = i - c * Ktot;
  float w = (k < Ka) ? Wa[(size_t)k * NOUT + c] : Wb[(size_t)(k - Ka) * NOUT + c];
  Bt[i] = f2bf(w);
}

// ---------------- eps mean over leading dim of 10 ----------------
__global__ void epsmean_kernel(const float* __restrict__ eps, float* __restrict__ em, int total) {
  int i = blockIdx.x * blockDim.x + threadIdx.x;
  if (i >= total) return;
  float s = 0.0f;
#pragma unroll
  for (int k = 0; k < 10; ++k) s += eps[(size_t)k * (size_t)total + i];
  em[i] = s * 0.1f;
}

// ---------------- z = mean + exp(log_std) * eps_mean, emitted as bf16 GEMM operand ----------------
__global__ void z_kernel(const float* __restrict__ mz, const float* __restrict__ em,
                         __bf16* __restrict__ zb, int N) {
  int i = blockIdx.x * blockDim.x + threadIdx.x;
  if (i >= N * DIM_OUT) return;
  int n = i >> 6;
  int o = i & 63;
  float mean = mz[(size_t)n * (2 * DIM_OUT) + o];
  float ls   = mz[(size_t)n * (2 * DIM_OUT) + DIM_OUT + o];
  zb[i] = f2bf(mean + expf(ls) * em[i]);
}

// ---------------- generic tall-skinny WMMA GEMM, bf16 in / f32 accumulate ----------------
// C[N x NOUT] = A[N x K] @ Bt^T + bias ; optional ReLU+BN epilogue ; optional residual add.
// Block = 256 threads (8 waves). Block owns 16 rows; wave w owns col tiles w, w+8, ...
// A fragment per lane: row = lane&15, k-base = 16*(lane>>4),
//   half0 = K[kb..kb+7], half1 = K[kb+8..kb+15]  -> two contiguous 16B loads.
// B fragment per lane (contiguous k order, same as A): col = lane&15, K range = 16*(lane>>4)..+15
//   -> one contiguous 32B load from the transposed weight.
__global__ void __launch_bounds__(256)
wmma_gemm(const __bf16* __restrict__ A, const __bf16* __restrict__ Bt,
          const float* __restrict__ bias, const float* __restrict__ gamma,
          const float* __restrict__ beta, const float* __restrict__ resid,
          float* __restrict__ C, int K, int NOUT) {
  const int lane = threadIdx.x & 31;
  const int wave = threadIdx.x >> 5;
  const int r    = lane & 15;
  const int hi   = lane >> 4;
  const int row0 = blockIdx.x << 4;
  const int ntile = NOUT >> 4;
  const __bf16* Arow = A + (size_t)(row0 + r) * (size_t)K;
  const float bnk = 0.99999499f;  // 1/sqrt(1 + 1e-5)

  for (int t = wave; t < ntile; t += 8) {
    const int col0 = t << 4;
    const __bf16* Bcol = Bt + (size_t)(col0 + r) * (size_t)K;
    v8f acc = {};
    for (int k0 = 0; k0 < K; k0 += 32) {
      union { v16bf v; v8bf h[2]; } a;
      a.h[0] = *(const v8bf*)(Arow + k0 + hi * 16);
      a.h[1] = *(const v8bf*)(Arow + k0 + hi * 16 + 8);
      v16bf b = *(const v16bf*)(Bcol + k0 + hi * 16);
      acc = __builtin_amdgcn_wmma_f32_16x16x32_bf16(
          /*neg_a=*/false, a.v, /*neg_b=*/false, b,
          /*c_mod=*/(short)0, acc, /*reuse_a=*/false, /*reuse_b=*/false);
    }
    const int col = col0 + r;
#pragma unroll
    for (int j = 0; j < 8; ++j) {
      const int m = j + (hi << 3);                 // D layout: VGPR j -> row j (+8 for hi half)
      const size_t idx = (size_t)(row0 + m) * (size_t)NOUT + col;
      float y = acc[j] + bias[col];
      if (gamma) {                                  // fused ReLU + BN(eval)
        y = fmaxf(y, 0.0f);
        y = y * (gamma[col] * bnk) + beta[col];
      }
      if (resid) y += resid[idx];
      C[idx] = y;
    }
  }
}

extern "C" void kernel_launch(void* const* d_in, const int* in_sizes, int n_in,
                              void* d_out, int out_size, void* d_ws, size_t ws_size,
                              hipStream_t stream) {
  (void)in_sizes; (void)n_in; (void)out_size; (void)ws_size;

  const float* x   = (const float*)d_in[0];
  const int*   ei  = (const int*)d_in[1];
  const float* eps = (const float*)d_in[2];
  const float* W1l = (const float*)d_in[3];
  const float* b1  = (const float*)d_in[4];
  const float* W1r = (const float*)d_in[5];
  const float* g1  = (const float*)d_in[6];
  const float* be1 = (const float*)d_in[7];
  const float* W2l = (const float*)d_in[8];
  const float* b2  = (const float*)d_in[9];
  const float* W2r = (const float*)d_in[10];
  const float* g2  = (const float*)d_in[11];
  const float* be2 = (const float*)d_in[12];
  const float* W3l = (const float*)d_in[13];
  const float* b3  = (const float*)d_in[14];
  const float* W3r = (const float*)d_in[15];
  const float* Wd1 = (const float*)d_in[16];
  const float* bd1 = (const float*)d_in[17];
  const float* g3  = (const float*)d_in[18];
  const float* be3 = (const float*)d_in[19];
  const float* Wd2 = (const float*)d_in[20];
  const float* bd2 = (const float*)d_in[21];
  float* out = (float*)d_out;

  const int* src = ei;
  const int* dst = ei + N_EDGES;

  // ---- carve workspace ----
  char* ws = (char*)d_ws;
  size_t off = 0;
  auto carve = [&](size_t bytes) -> char* {
    char* p = ws + off;
    off += (bytes + 255) & ~(size_t)255;
    return p;
  };
  float*  deg = (float*)carve((size_t)N_NODES * 4);
  float*  agg = (float*)carve((size_t)N_NODES * DIM_IN * 4);        // reused every layer; aliased by em later
  float*  x1  = (float*)carve((size_t)N_NODES * DIM_HID * 4);
  float*  x2  = (float*)carve((size_t)N_NODES * 2 * DIM_HID * 4);
  float*  mz  = (float*)carve((size_t)N_NODES * DIM_HID * 4);       // aliased as h after z_kernel consumes it
  __bf16* Abf = (__bf16*)carve((size_t)N_NODES * 512 * 2);          // bf16 A operand (max K=512)
  __bf16* Bt  = (__bf16*)carve((size_t)256 * 512 * 2);              // bf16 transposed weight
  float*  em  = agg;   // agg is dead after layer-3 A build
  float*  h   = mz;    // mz is dead after z_kernel

  auto nblk = [](long long total, int bs) { return (unsigned)((total + bs - 1) / bs); };

  // ---- degree (shared by all 3 SAGE layers) ----
  hipMemsetAsync(deg, 0, (size_t)N_NODES * 4, stream);
  deg_kernel<<<nblk(N_EDGES, 256), 256, 0, stream>>>(dst, deg, N_EDGES);

  // ---- SAGE layer 1: x1 = bn(relu([mean_agg(x)|x] @ [W1l;W1r] + b1)) ----
  hipMemsetAsync(agg, 0, (size_t)N_NODES * DIM_IN * 4, stream);
  scatter_kernel<<<nblk((long long)N_EDGES * (DIM_IN / 4), 256), 256, 0, stream>>>(x, src, dst, agg, N_EDGES, DIM_IN);
  build_a_kernel<<<nblk((long long)N_NODES * DIM_IN, 256), 256, 0, stream>>>(agg, deg, x, Abf, N_NODES, DIM_IN);
  prep_w_kernel<<<nblk((long long)DIM_HID * 512, 256), 256, 0, stream>>>(W1l, W1r, Bt, DIM_IN, DIM_IN, DIM_HID);
  wmma_gemm<<<N_NODES / 16, 256, 0, stream>>>(Abf, Bt, b1, g1, be1, nullptr, x1, 2 * DIM_IN, DIM_HID);

  // ---- SAGE layer 2: x2 = bn(relu([mean_agg(x1)|x1] @ [W2l;W2r] + b2)) ----
  hipMemsetAsync(agg, 0, (size_t)N_NODES * DIM_HID * 4, stream);
  scatter_kernel<<<nblk((long long)N_EDGES * (DIM_HID / 4), 256), 256, 0, stream>>>(x1, src, dst, agg, N_EDGES, DIM_HID);
  build_a_kernel<<<nblk((long long)N_NODES * DIM_HID, 256), 256, 0, stream>>>(agg, deg, x1, Abf, N_NODES, DIM_HID);
  prep_w_kernel<<<nblk((long long)(2 * DIM_HID) * (2 * DIM_HID), 256), 256, 0, stream>>>(W2l, W2r, Bt, DIM_HID, DIM_HID, 2 * DIM_HID);
  wmma_gemm<<<N_NODES / 16, 256, 0, stream>>>(Abf, Bt, b2, g2, be2, nullptr, x2, 2 * DIM_HID, 2 * DIM_HID);

  // ---- SAGE layer 3 (no relu/bn): mz = [mean_agg(x2)|x2] @ [W3l;W3r] + b3 ----
  hipMemsetAsync(agg, 0, (size_t)N_NODES * 2 * DIM_HID * 4, stream);
  scatter_kernel<<<nblk((long long)N_EDGES * (2 * DIM_HID / 4), 256), 256, 0, stream>>>(x2, src, dst, agg, N_EDGES, 2 * DIM_HID);
  build_a_kernel<<<nblk((long long)N_NODES * 2 * DIM_HID, 256), 256, 0, stream>>>(agg, deg, x2, Abf, N_NODES, 2 * DIM_HID);
  prep_w_kernel<<<nblk((long long)(2 * DIM_OUT) * 512, 256), 256, 0, stream>>>(W3l, W3r, Bt, 2 * DIM_HID, 2 * DIM_HID, 2 * DIM_OUT);
  wmma_gemm<<<N_NODES / 16, 256, 0, stream>>>(Abf, Bt, b3, nullptr, nullptr, nullptr, mz, 512, 2 * DIM_OUT);

  // ---- reparameterization: zb = bf16(mean + exp(log_std) * mean(eps)) ----
  epsmean_kernel<<<nblk((long long)N_NODES * DIM_OUT, 256), 256, 0, stream>>>(eps, em, N_NODES * DIM_OUT);
  z_kernel<<<nblk((long long)N_NODES * DIM_OUT, 256), 256, 0, stream>>>(mz, em, Abf, N_NODES);

  // ---- decoder 1 (fused residual): h = bn(relu(z @ Wd1 + bd1)) + x1 ----
  prep_w_kernel<<<nblk((long long)DIM_HID * DIM_OUT, 256), 256, 0, stream>>>(Wd1, nullptr, Bt, DIM_OUT, 0, DIM_HID);
  wmma_gemm<<<N_NODES / 16, 256, 0, stream>>>(Abf, Bt, bd1, g3, be3, x1, h, DIM_OUT, DIM_HID);

  // ---- decoder 2: out = h @ Wd2 + bd2 ----
  conv_bf_kernel<<<nblk((long long)N_NODES * DIM_HID, 256), 256, 0, stream>>>(h, Abf, N_NODES * DIM_HID);
  prep_w_kernel<<<nblk((long long)DIM_IN * DIM_HID, 256), 256, 0, stream>>>(Wd2, nullptr, Bt, DIM_HID, 0, DIM_IN);
  wmma_gemm<<<N_NODES / 16, 256, 0, stream>>>(Abf, Bt, bd2, nullptr, nullptr, nullptr, out, DIM_HID, DIM_IN);
}
